// Top_Down_Baseline_63462436765910
// MI455X (gfx1250) — matrix operand order repaired and checked by reading, not compiled, not run on hardware
//
#include <hip/hip_runtime.h>
#include <math.h>

// ---------------------------------------------------------------- types ----
typedef __attribute__((ext_vector_type(16))) __bf16 v16bf;
typedef __attribute__((ext_vector_type(8)))  float  v8f;

union FragBF { v16bf v; uint4 q[2]; };

__device__ __forceinline__ unsigned short f2bf(float f) {
    unsigned int u = __float_as_uint(f);
    unsigned int r = u + 0x7fffu + ((u >> 16) & 1u);   // round-to-nearest-even
    return (unsigned short)(r >> 16);
}

// ---------------------------------------------------------------- GEMM -----
// C[M,N] = act(A[M,K] @ W[K,N] + bias[N]); A,W,bias,C fp32 in HBM,
// bf16 in LDS, fp32 WMMA accumulation.  Block tile 128x128, BK=32.
// Register-staged pipeline: global loads of tile k+1 overlap WMMA of tile k.
// B fragments are consumed one at a time to keep VGPR count <= 256.
#define BM 128
#define BN 128
#define BKK 32
#define LDT 40   // LDS row stride in ushort (80B, multiple of 16B)

__global__ __launch_bounds__(256)
void gemm_bf16_wmma(const float* __restrict__ A, const float* __restrict__ W,
                    const float* __restrict__ bias, float* __restrict__ Cmat,
                    int M, int N, int K, int do_relu)
{
    __shared__ __align__(16) unsigned short sA[BM * LDT];
    __shared__ __align__(16) unsigned short sB[BN * LDT];

    const int tid   = threadIdx.x;
    const int wave  = tid >> 5;
    const int lane  = tid & 31;
    const int wm    = wave & 3;      // 4 waves along M: 32 rows each
    const int wn    = wave >> 2;     // 2 waves along N: 64 cols each
    const int blockM = blockIdx.y * BM;
    const int blockN = blockIdx.x * BN;
    const int lhalf = lane >> 4;
    const int l16   = lane & 15;

    const bool vecA = (blockM + BM <= M) && ((K & 3) == 0);
    const bool vecB = (blockN + BN <= N) && ((N & 3) == 0);

    float rA[16], rB[16];

    // ---- global -> registers (fast: b128 vector loads; slow: guarded) ----
    auto loadA = [&](int k0) {
        if (vecA && (k0 + BKK <= K)) {
            #pragma unroll
            for (int i = 0; i < 4; ++i) {
                int idx = i * 256 + tid;
                int r = idx >> 3;            // row in tile
                int c = (idx & 7) * 4;       // col in tile (mult of 4)
                const float4 v = *(const float4*)&A[(size_t)(blockM + r) * K + k0 + c];
                rA[4*i+0] = v.x; rA[4*i+1] = v.y; rA[4*i+2] = v.z; rA[4*i+3] = v.w;
            }
        } else {
            #pragma unroll
            for (int i = 0; i < 16; ++i) {
                int idx = i * 256 + tid;
                int r = idx >> 5, c = idx & 31;
                int gr = blockM + r, gc = k0 + c;
                rA[i] = (gr < M && gc < K) ? A[(size_t)gr * K + gc] : 0.f;
            }
        }
    };
    auto loadB = [&](int k0) {
        if (vecB && (k0 + BKK <= K)) {
            #pragma unroll
            for (int i = 0; i < 4; ++i) {
                int idx = i * 256 + tid;
                int k = idx >> 5;            // k in tile
                int n = (idx & 31) * 4;      // n in tile (mult of 4)
                const float4 v = *(const float4*)&W[(size_t)(k0 + k) * N + blockN + n];
                rB[4*i+0] = v.x; rB[4*i+1] = v.y; rB[4*i+2] = v.z; rB[4*i+3] = v.w;
            }
        } else {
            #pragma unroll
            for (int i = 0; i < 16; ++i) {
                int idx = i * 256 + tid;
                int k = idx >> 7, n = idx & 127;
                int gk = k0 + k, gn = blockN + n;
                rB[i] = (gk < K && gn < N) ? W[(size_t)gk * N + gn] : 0.f;
            }
        }
    };
    // ---- registers -> LDS (bf16), mapping matches the load path ----
    auto storeA = [&](int k0) {
        if (vecA && (k0 + BKK <= K)) {
            #pragma unroll
            for (int i = 0; i < 4; ++i) {
                int idx = i * 256 + tid;
                int r = idx >> 3;
                int c = (idx & 7) * 4;
                uint2 p;
                p.x = ((unsigned)f2bf(rA[4*i+1]) << 16) | f2bf(rA[4*i+0]);
                p.y = ((unsigned)f2bf(rA[4*i+3]) << 16) | f2bf(rA[4*i+2]);
                *(uint2*)&sA[r * LDT + c] = p;      // ds_store_b64
            }
        } else {
            #pragma unroll
            for (int i = 0; i < 16; ++i) {
                int idx = i * 256 + tid;
                int r = idx >> 5, c = idx & 31;
                sA[r * LDT + c] = f2bf(rA[i]);
            }
        }
    };
    auto storeB = [&](int k0) {
        if (vecB && (k0 + BKK <= K)) {
            #pragma unroll
            for (int i = 0; i < 4; ++i) {
                int idx = i * 256 + tid;
                int k = idx >> 5;
                int n = (idx & 31) * 4;
                sB[(n+0) * LDT + k] = f2bf(rB[4*i+0]);   // transposed: sB[n][k]
                sB[(n+1) * LDT + k] = f2bf(rB[4*i+1]);
                sB[(n+2) * LDT + k] = f2bf(rB[4*i+2]);
                sB[(n+3) * LDT + k] = f2bf(rB[4*i+3]);
            }
        } else {
            #pragma unroll
            for (int i = 0; i < 16; ++i) {
                int idx = i * 256 + tid;
                int k = idx >> 7, n = idx & 127;
                sB[n * LDT + k] = f2bf(rB[i]);
            }
        }
    };

    v8f acc[2][4];
    for (int i = 0; i < 2; ++i)
        for (int j = 0; j < 4; ++j)
            for (int e = 0; e < 8; ++e) acc[i][j][e] = 0.f;

    const int ntiles = (K + BKK - 1) / BKK;
    loadA(0); loadB(0);

    for (int kt = 0; kt < ntiles; ++kt) {
        const int k0 = kt * BKK;
        storeA(k0); storeB(k0);
        __syncthreads();
        if (kt + 1 < ntiles) { loadA(k0 + BKK); loadB(k0 + BKK); }  // overlap w/ WMMA

        FragBF a[2];
        #pragma unroll
        for (int tm = 0; tm < 2; ++tm) {
            int m = wm * 32 + tm * 16 + l16;
            a[tm].q[0] = *(const uint4*)&sA[m * LDT + lhalf * 8];
            a[tm].q[1] = *(const uint4*)&sA[m * LDT + 16 + lhalf * 8];
        }
        // one live B fragment at a time (keeps VGPRs <= 256)
        #pragma unroll
        for (int tn = 0; tn < 4; ++tn) {
            FragBF b;
            int n = wn * 64 + tn * 16 + l16;
            b.q[0] = *(const uint4*)&sB[n * LDT + lhalf * 8];
            b.q[1] = *(const uint4*)&sB[n * LDT + 16 + lhalf * 8];
            acc[0][tn] = __builtin_amdgcn_wmma_f32_16x16x32_bf16(
                false, a[0].v, false, b.v, (short)0, acc[0][tn], false, false);
            acc[1][tn] = __builtin_amdgcn_wmma_f32_16x16x32_bf16(
                false, a[1].v, false, b.v, (short)0, acc[1][tn], false, false);
        }
        __syncthreads();
    }

    // epilogue: C/D layout — VGPR r: M = r + 8*(lane/16), N = lane%16
    for (int tm = 0; tm < 2; ++tm) {
        int mbase = blockM + wm * 32 + tm * 16 + lhalf * 8;
        for (int tn = 0; tn < 4; ++tn) {
            int col = blockN + wn * 64 + tn * 16 + l16;
            if (col >= N) continue;
            float bval = bias ? bias[col] : 0.f;
            #pragma unroll
            for (int r = 0; r < 8; ++r) {
                int row = mbase + r;
                if (row < M) {
                    float v = acc[tm][tn][r] + bval;
                    if (do_relu) v = fmaxf(v, 0.f);
                    Cmat[(size_t)row * N + col] = v;
                }
            }
        }
    }
}

// -------------------------------------------------- small fused kernels ----
__global__ void k_transpose_vorg(const float* __restrict__ in,
                                 float* __restrict__ out,
                                 int Bn, int Cn, int KK)
{
    size_t idx = (size_t)blockIdx.x * blockDim.x + threadIdx.x;
    size_t total = (size_t)Bn * Cn * KK;
    if (idx >= total) return;
    int c = (int)(idx % Cn);
    size_t t = idx / Cn;
    int k = (int)(t % KK);
    int b = (int)(t / KK);
    out[((size_t)b * KK + k) * Cn + c] = in[((size_t)b * Cn + c) * KK + k];
}

__global__ void k_build_rv(const int* __restrict__ gt_verb,
                           const int* __restrict__ role_idx,
                           const float* __restrict__ vt,
                           const float* __restrict__ rt,
                           float* __restrict__ rv, int Bn, int R, int E)
{
    size_t idx = (size_t)blockIdx.x * blockDim.x + threadIdx.x;
    size_t total = (size_t)Bn * R * 2 * E;
    if (idx >= total) return;
    int e  = (int)(idx % (2 * E));
    int br = (int)(idx / (2 * E));
    int b = br / R, r = br % R;
    float v;
    if (e < E) v = vt[(size_t)gt_verb[b] * E + e];
    else       v = rt[(size_t)role_idx[b * R + r] * E + (e - E)];
    rv[idx] = v;
}

// softmax attention over K positions, fused with weighted value sum.
// grid = BR blocks, 256 threads (8 wave32s).
__global__ __launch_bounds__(256)
void k_vatt(const float* __restrict__ V,    // [B,K,H]  relu(img@av_W+b)
            const float* __restrict__ img,  // [B,K,C]
            const float* __restrict__ q,    // [BR,H]   relu(q@aq_W+b)
            const float* __restrict__ aoW,  // [H]
            const float* __restrict__ aoB,  // [1]
            float* __restrict__ vemb,       // [BR,C]
            int R, int K49, int C, int H)
{
    __shared__ float s_att[64];
    int br = blockIdx.x, b = br / R;
    int w = threadIdx.x >> 5, ln = threadIdx.x & 31;
    const float* qrow = q + (size_t)br * H;

    for (int k = w; k < K49; k += 8) {
        const float* vrow = V + ((size_t)b * K49 + k) * H;
        float acc = 0.f;
        for (int h = ln; h < H; h += 32) acc += vrow[h] * qrow[h] * aoW[h];
        for (int off = 16; off > 0; off >>= 1) acc += __shfl_xor(acc, off, 32);
        if (ln == 0) s_att[k] = acc + aoB[0];
    }
    __syncthreads();
    if (threadIdx.x == 0) {
        float mx = -1e30f;
        for (int k = 0; k < K49; ++k) mx = fmaxf(mx, s_att[k]);
        float sum = 0.f;
        for (int k = 0; k < K49; ++k) { float e = __expf(s_att[k] - mx); s_att[k] = e; sum += e; }
        float inv = 1.f / sum;
        for (int k = 0; k < K49; ++k) s_att[k] *= inv;
    }
    __syncthreads();
    for (int c = threadIdx.x; c < C; c += blockDim.x) {
        const float* ib = img + (size_t)b * K49 * C + c;
        float acc = 0.f;
        for (int k = 0; k < K49; ++k) acc += s_att[k] * ib[(size_t)k * C];
        vemb[(size_t)br * C + c] = acc;
    }
}

// MFB: signed-sqrt + L2 normalize per row.  grid = rows, 256 thr, H = 1024.
__global__ __launch_bounds__(256)
void k_mfb(const float* __restrict__ q, const float* __restrict__ v,
           float* __restrict__ out, int H)
{
    __shared__ float red[256];
    int row = blockIdx.x, tid = threadIdx.x;
    const float* qp = q + (size_t)row * H;
    const float* vp = v + (size_t)row * H;
    float s[4]; float ss = 0.f;
    #pragma unroll
    for (int i = 0; i < 4; ++i) {
        int h = tid + i * 256;
        float z = qp[h] * vp[h];
        float t = (z >= 0.f) ? sqrtf(z) : -sqrtf(-z);
        s[i] = t; ss += t * t;
    }
    red[tid] = ss; __syncthreads();
    for (int st = 128; st > 0; st >>= 1) { if (tid < st) red[tid] += red[tid + st]; __syncthreads(); }
    float inv = 1.f / fmaxf(sqrtf(red[0]), 1e-12f);
    #pragma unroll
    for (int i = 0; i < 4; ++i) out[(size_t)row * H + tid + i * 256] = s[i] * inv;
}

// masked 6x6 graph attention per batch element.  grid = B, 256 threads.
__global__ __launch_bounds__(256)
void k_graph(const float* __restrict__ qh, const float* __restrict__ kh,
             const float* __restrict__ vh, const int* __restrict__ mask,
             float* __restrict__ neigh, int R, int H)
{
    __shared__ float sc[6][6];
    __shared__ float at[6][6];
    int b = blockIdx.x;
    int w = threadIdx.x >> 5, ln = threadIdx.x & 31;
    float scale = 1.f / sqrtf((float)H);

    for (int p = w; p < R * R; p += 8) {
        int i = p / R, j = p % R;
        const float* qp = qh + ((size_t)b * R + i) * H;
        const float* kp = kh + ((size_t)b * R + j) * H;
        float acc = 0.f;
        for (int h = ln; h < H; h += 32) acc += qp[h] * kp[h];
        for (int off = 16; off > 0; off >>= 1) acc += __shfl_xor(acc, off, 32);
        if (ln == 0) sc[i][j] = acc * scale;
    }
    __syncthreads();
    if (threadIdx.x < (unsigned)R) {
        int i = threadIdx.x;
        float vals[6]; float mx = -1e30f;
        for (int j = 0; j < R; ++j) {
            float m = (float)mask[(size_t)b * R * R + i * R + j] * ((i == j) ? 0.f : 1.f);
            float sv = (m > 0.f) ? sc[i][j] : -1e9f;
            vals[j] = sv; mx = fmaxf(mx, sv);
        }
        float sum = 0.f;
        for (int j = 0; j < R; ++j) { float e = __expf(vals[j] - mx); vals[j] = e; sum += e; }
        float inv = 1.f / sum;
        for (int j = 0; j < R; ++j) at[i][j] = vals[j] * inv;
    }
    __syncthreads();
    for (int h = threadIdx.x; h < H; h += blockDim.x) {
        float vv[6];
        for (int j = 0; j < R; ++j) vv[j] = vh[((size_t)b * R + j) * H + h];
        for (int i = 0; i < R; ++i) {
            float a = 0.f;
            for (int j = 0; j < R; ++j) a += at[i][j] * vv[j];
            neigh[((size_t)b * R + i) * H + h] = a;
        }
    }
}

__global__ void k_concat(const float* __restrict__ neigh,
                         const float* __restrict__ rv,
                         float* __restrict__ cat2, int BR, int H, int E2)
{
    size_t idx = (size_t)blockIdx.x * blockDim.x + threadIdx.x;
    size_t total = (size_t)BR * (H + E2);
    if (idx >= total) return;
    int e  = (int)(idx % (H + E2));
    int br = (int)(idx / (H + E2));
    cat2[idx] = (e < H) ? neigh[(size_t)br * H + e] : rv[(size_t)br * E2 + (e - H)];
}

// out2 <- (1-g)*out_verb + g*tanh(out2+ans0), g = sigmoid(out2+out_verb)
__global__ void k_gate(float* __restrict__ out2,
                       const float* __restrict__ out_verb,
                       const float* __restrict__ ans0, size_t total)
{
    size_t idx = (size_t)blockIdx.x * blockDim.x + threadIdx.x;
    if (idx >= total) return;
    float o2 = out2[idx], ov = out_verb[idx], a0 = ans0[idx];
    float g = 1.f / (1.f + __expf(-(o2 + ov)));
    out2[idx] = (1.f - g) * ov + g * tanhf(o2 + a0);
}

// --------------------------------------------------------------- driver ----
extern "C" void kernel_launch(void* const* d_in, const int* in_sizes, int n_in,
                              void* d_out, int out_size, void* d_ws, size_t ws_size,
                              hipStream_t stream)
{
    (void)in_sizes; (void)n_in; (void)out_size;
    constexpr int Bn = 256, Rr = 6, K49 = 49, Cc = 512, Hh = 1024, Ee = 300, NLc = 2001;
    constexpr int BR = Bn * Rr;          // 1536
    constexpr int BK49 = Bn * K49;       // 12544
    constexpr int E2 = 2 * Ee;           // 600
    constexpr int CAT = Hh + E2;         // 1624

    const float* v_org    = (const float*)d_in[0];
    const float* img_feat = (const float*)d_in[1];
    const int*   gt_verb  = (const int*)d_in[2];
    const int*   role_idx = (const int*)d_in[3];
    const int*   maskp    = (const int*)d_in[4];
    const float* verb_tab = (const float*)d_in[5];
    const float* role_tab = (const float*)d_in[6];
    const float* qc_W = (const float*)d_in[7];  const float* qc_b = (const float*)d_in[8];
    const float* av_W = (const float*)d_in[9];  const float* av_b = (const float*)d_in[10];
    const float* aq_W = (const float*)d_in[11]; const float* aq_b = (const float*)d_in[12];
    const float* ao_W = (const float*)d_in[13]; const float* ao_b = (const float*)d_in[14];
    const float* vn_W = (const float*)d_in[15]; const float* vn_b = (const float*)d_in[16];
    const float* qn_W = (const float*)d_in[17]; const float* qn_b = (const float*)d_in[18];
    const float* Wq   = (const float*)d_in[19]; const float* Wk   = (const float*)d_in[20];
    const float* Wv   = (const float*)d_in[21]; const float* Wo   = (const float*)d_in[22];
    const float* uqc_W = (const float*)d_in[23]; const float* uqc_b = (const float*)d_in[24];
    const float* cls_W = (const float*)d_in[25]; const float* cls_b = (const float*)d_in[26];
    float* logits = (float*)d_out;

    // ---- workspace carve-out (256B aligned) ----
    char* ws = (char*)d_ws; size_t off = 0;
    auto alloc = [&](size_t nfloat) -> float* {
        float* p = (float*)(ws + off);
        off += ((nfloat * sizeof(float)) + 255) & ~(size_t)255;
        return p;
    };
    float* img0     = alloc((size_t)BK49 * Cc);   // [B,49,512]  v_org transposed
    float* rv       = alloc((size_t)BR * E2);     // [BR,600]
    float* q_emb    = alloc((size_t)BR * Hh);
    float* V1       = alloc((size_t)BK49 * Hh);   // relu(img0@av_W+b)   (att1/att2)
    float* Vv       = alloc((size_t)BK49 * Hh);   // relu(img_feat@av_W+b)
    float* qA       = alloc((size_t)BR * Hh);     // reused as qA2
    float* v_emb    = alloc((size_t)BR * Cc);     // reused as v_emb2
    float* v_embv   = alloc((size_t)BR * Cc);
    float* vr       = alloc((size_t)BR * Hh);     // v_repr / v_repr_verb / v_repr2
    float* q_repr   = alloc((size_t)BR * Hh);     // q_repr / q_repr2
    float* ans0     = alloc((size_t)BR * Hh);
    float* out_verb = alloc((size_t)BR * Hh);
    float* qh       = alloc((size_t)BR * Hh);
    float* kh       = alloc((size_t)BR * Hh);
    float* vh       = alloc((size_t)BR * Hh);
    float* neigh_p  = alloc((size_t)BR * Hh);
    float* neigh    = alloc((size_t)BR * Hh);
    float* cat2     = alloc((size_t)BR * CAT);
    float* uq       = alloc((size_t)BR * Hh);
    float* out2     = alloc((size_t)BR * Hh);     // becomes out_f in-place
    if (off > ws_size) return;                    // workspace too small

    auto gemm = [&](const float* A, const float* W, const float* bias,
                    float* Cout, int M, int N, int Kd, int relu) {
        dim3 g((N + BN - 1) / BN, (M + BM - 1) / BM);
        gemm_bf16_wmma<<<g, 256, 0, stream>>>(A, W, bias, Cout, M, N, Kd, relu);
    };
    auto eblocks = [](size_t n) { return (unsigned)((n + 255) / 256); };

    // 1. img0[b,k,c] = v_org[b,c,k]
    k_transpose_vorg<<<eblocks((size_t)BK49 * Cc), 256, 0, stream>>>(v_org, img0, Bn, Cc, K49);
    // 2. rv = concat(verb_e, role_e)
    k_build_rv<<<eblocks((size_t)BR * E2), 256, 0, stream>>>(gt_verb, role_idx, verb_tab, role_tab, rv, Bn, Rr, Ee);
    // 3. q_emb = relu(rv @ qc_W + qc_b)
    gemm(rv, qc_W, qc_b, q_emb, BR, Hh, E2, 1);
    // 4. V1 = relu(img0 @ av_W + av_b)   (per-b, shared across R and att1/att2)
    gemm(img0, av_W, av_b, V1, BK49, Hh, Cc, 1);
    // 5. Vv = relu(img_feat @ av_W + av_b)
    gemm(img_feat, av_W, av_b, Vv, BK49, Hh, Cc, 1);
    // 6. qA = relu(q_emb @ aq_W + aq_b)  (shared by att1/att_v)
    gemm(q_emb, aq_W, aq_b, qA, BR, Hh, Hh, 1);
    // 7/8. attention pools
    k_vatt<<<BR, 256, 0, stream>>>(V1, img0,     qA, ao_W, ao_b, v_emb,  Rr, K49, Cc, Hh);
    k_vatt<<<BR, 256, 0, stream>>>(Vv, img_feat, qA, ao_W, ao_b, v_embv, Rr, K49, Cc, Hh);
    // 9/10. v_repr, q_repr
    gemm(v_emb, vn_W, vn_b, vr,     BR, Hh, Cc, 1);
    gemm(q_emb, qn_W, qn_b, q_repr, BR, Hh, Hh, 1);
    // 11. ans0 = mfb(q_repr, v_repr)
    k_mfb<<<BR, 256, 0, stream>>>(q_repr, vr, ans0, Hh);
    // 12/13. out_verb = mfb(q_repr, relu(v_embv@vn_W+b))
    gemm(v_embv, vn_W, vn_b, vr, BR, Hh, Cc, 1);
    k_mfb<<<BR, 256, 0, stream>>>(q_repr, vr, out_verb, Hh);
    // 14-16. graph attention over roles
    gemm(ans0, Wq, nullptr, qh, BR, Hh, Hh, 0);
    gemm(ans0, Wk, nullptr, kh, BR, Hh, Hh, 0);
    gemm(ans0, Wv, nullptr, vh, BR, Hh, Hh, 0);
    k_graph<<<Bn, 256, 0, stream>>>(qh, kh, vh, maskp, neigh_p, Rr, Hh);
    gemm(neigh_p, Wo, nullptr, neigh, BR, Hh, Hh, 0);
    // 17/18. uq = relu(concat(neigh, rv) @ uqc_W + uqc_b)
    k_concat<<<eblocks((size_t)BR * CAT), 256, 0, stream>>>(neigh, rv, cat2, BR, Hh, E2);
    gemm(cat2, uqc_W, uqc_b, uq, BR, Hh, CAT, 1);
    // 19/20. att2 (reuses V1/img0)
    gemm(uq, aq_W, aq_b, qA, BR, Hh, Hh, 1);
    k_vatt<<<BR, 256, 0, stream>>>(V1, img0, qA, ao_W, ao_b, v_emb, Rr, K49, Cc, Hh);
    // 21-23. out2 = mfb(relu(uq@qn_W+b), relu(v_emb2@vn_W+b))
    gemm(v_emb, vn_W, vn_b, vr,     BR, Hh, Cc, 1);
    gemm(uq,    qn_W, qn_b, q_repr, BR, Hh, Hh, 1);
    k_mfb<<<BR, 256, 0, stream>>>(q_repr, vr, out2, Hh);
    // 24. gate -> out_f (in place in out2)
    k_gate<<<eblocks((size_t)BR * Hh), 256, 0, stream>>>(out2, out_verb, ans0, (size_t)BR * Hh);
    // 25. logits = out_f @ cls_W + cls_b
    gemm(out2, cls_W, cls_b, logits, BR, NLc, Hh, 0);
}
